// Combine_16621523436266
// MI455X (gfx1250) — matrix operand order repaired and checked by reading, not compiled, not run on hardware
//
#include <hip/hip_runtime.h>

// ---------------------------------------------------------------------------
// MoE dynamic conv:  out[b] = conv3x3(x[b], sum_e att[b,e]*W[e]) + mixed bias
// B=32, C=256, O=256, H=W=56, K=3, E=8
// Implicit GEMM, v_wmma_f32_16x16x32_bf16, M=O=256, N=HW=3136, K=C*9=2304.
// Double-buffered LDS pipeline, K-step 64, pre-padded bf16 input (58x58).
// A-tile staged with GLOBAL_LOAD_ASYNC_TO_LDS_B128 (ASYNCcnt) when available.
// ---------------------------------------------------------------------------

typedef __attribute__((ext_vector_type(16))) __bf16 v16bf;
typedef __attribute__((ext_vector_type(8)))  __bf16 v8bf;
typedef __attribute__((ext_vector_type(8)))  float  v8f;
typedef __attribute__((ext_vector_type(4)))  int    v4i;

#if __has_builtin(__builtin_amdgcn_global_load_async_to_lds_b128) && \
    __has_builtin(__builtin_amdgcn_s_wait_asynccnt)
#define USE_ASYNC_LDS 1
#else
#define USE_ASYNC_LDS 0
#endif

#define NB 32
#define CI 256
#define CO 256
#define HH 56
#define WW 56
#define HW 3136            // 56*56
#define PH 58
#define PHW 3364           // 58*58
#define NE 8
#define NIT 36             // 9 taps * (256/64) channel chunks

// ---------------------------------------------------------------------------
// Kernel 1: per-sample combined weights, bf16, layout Wc[b][tap][o][c]
// ---------------------------------------------------------------------------
__global__ __launch_bounds__(256) void combine_w_kernel(
    const float* __restrict__ att, const float* __restrict__ w,
    __bf16* __restrict__ wc) {
  long idx = (long)blockIdx.x * 256 + threadIdx.x;
  int c  = (int)(idx & 255);
  int o  = (int)((idx >> 8) & 255);
  int kv = (int)((idx >> 16) % 9);
  int b  = (int)(idx / 589824);   // 9*256*256
  float s = 0.0f;
#pragma unroll
  for (int e = 0; e < NE; ++e)
    s += att[b * NE + e] * w[(((long)e * CO + o) * CI + c) * 9 + kv];
  wc[(((long)b * 9 + kv) * CO + o) * CI + c] = (__bf16)s;
}

// ---------------------------------------------------------------------------
// Kernel 2: per-sample combined bias (fp32)
// ---------------------------------------------------------------------------
__global__ __launch_bounds__(256) void combine_bias_kernel(
    const float* __restrict__ att, const float* __restrict__ bias,
    float* __restrict__ bc) {
  int o = threadIdx.x;
  int b = blockIdx.x;
  float s = 0.0f;
#pragma unroll
  for (int e = 0; e < NE; ++e) s += att[b * NE + e] * bias[e * CO + o];
  bc[b * CO + o] = s;
}

// ---------------------------------------------------------------------------
// Kernel 3: fp32 -> bf16 with zero halo: Xp[b][c][58][58]
// ---------------------------------------------------------------------------
__global__ __launch_bounds__(256) void convert_pad_kernel(
    const float* __restrict__ x, __bf16* __restrict__ xp) {
  long idx = (long)blockIdx.x * 256 + threadIdx.x;
  if (idx >= (long)NB * CI * PHW) return;
  int pw = (int)(idx % PH);
  int ph = (int)((idx / PH) % PH);
  long bc = idx / PHW;
  __bf16 v = (__bf16)0.0f;
  if (ph >= 1 && ph <= HH && pw >= 1 && pw <= WW)
    v = (__bf16)x[bc * HW + (ph - 1) * WW + (pw - 1)];
  xp[idx] = v;
}

// ---------------------------------------------------------------------------
// Kernel 4: implicit-GEMM conv, double-buffered, K-step 64
// grid = (49, 2, 32), block = 256 (8 wave32); block tile 128(M) x 64(N)
// ---------------------------------------------------------------------------
__global__ __launch_bounds__(256) void moe_conv_wmma_kernel(
    const __bf16* __restrict__ Wc, const __bf16* __restrict__ Xp,
    const float* __restrict__ Bc, float* __restrict__ out) {
  __shared__ __bf16 As[2][128][72];   // [buf][M][K=64], padded stride
  __shared__ __bf16 Bs[2][64][72];    // [buf][N][K=64] (N-major)

  const int tid  = threadIdx.x;
  const int lane = tid & 31;
  const int wave = tid >> 5;
  const int wm   = wave & 3;          // 4 waves along M
  const int wn   = wave >> 2;         // 2 waves along N
  const int n0   = blockIdx.x * 64;   // 49*64 == 3136
  const int m0   = blockIdx.y * 128;
  const int b    = blockIdx.z;

  // staging roles
  const int aRow = tid >> 1;            // A row (o), 0..127
  const int aCol = (tid & 1) * 32;      // K half, 64 bytes per thread
  const int bC   = tid >> 2;            // channel 0..63 (quad-coalesced n)
  const int bN   = (tid & 3) * 16;      // 16 pixels per thread

  // per-thread padded-image offsets for the 16 pixels (computed once)
  int q[16];
#pragma unroll
  for (int j = 0; j < 16; ++j) {
    int n = n0 + bN + j;
    q[j] = (n / WW) * PH + (n % WW);
  }

  const __bf16* wBase = Wc + (((long)b * 9) * CO + m0) * CI;
  const __bf16* xBase = Xp + (long)b * CI * PHW;

  v8f acc[2][2] = {};
  __bf16 br[16];
#if !USE_ASYNC_LDS
  v8bf ar[4];
#endif

  // ---- prologue: stage iteration 0 (kv=0 -> tap offset d=0, c0=0)
  {
    const __bf16* src = wBase + (long)aRow * CI + aCol;
#if USE_ASYNC_LDS
#pragma unroll
    for (int k = 0; k < 4; ++k)
      __builtin_amdgcn_global_load_async_to_lds_b128(
          (v4i*)(src + k * 8), (v4i*)&As[0][aRow][aCol + k * 8], 0, 0);
#else
#pragma unroll
    for (int k = 0; k < 4; ++k) ar[k] = *(const v8bf*)(src + k * 8);
#pragma unroll
    for (int k = 0; k < 4; ++k) *(v8bf*)&As[0][aRow][aCol + k * 8] = ar[k];
#endif
    const __bf16* xs = xBase + (long)bC * PHW;
#pragma unroll
    for (int j = 0; j < 16; ++j) br[j] = xs[q[j]];
#pragma unroll
    for (int j = 0; j < 16; ++j) Bs[0][bN + j][bC] = br[j];
  }
#if USE_ASYNC_LDS
  __builtin_amdgcn_s_wait_asynccnt(0);
#endif
  __syncthreads();

  const int mBase = wm * 32 + (lane & 15);
  const int nBase = wn * 32 + (lane & 15);
  const int ak0   = (lane >> 4) * 8;   // A: lanes<16 K{0..7,16..23}; >=16 K{8..15,24..31}
  const int bk0   = (lane >> 4) * 8;   // B: lanes<16 K{0..7,16..23}; >=16 K{8..15,24..31}

  for (int it = 0; it < NIT; ++it) {
    const int cur = it & 1;
    const int nxt = cur ^ 1;

    // ---- issue loads for stage it+1 (latency hidden behind WMMAs)
    if (it + 1 < NIT) {
      const int it1 = it + 1;
      const int kv  = it1 >> 2;
      const int c0  = (it1 & 3) * 64;
      const int d   = (kv / 3) * PH + (kv % 3);
      const __bf16* src = wBase + (long)kv * (CO * CI) + (long)aRow * CI + c0 + aCol;
#if USE_ASYNC_LDS
      // A tile: DMA directly into the other LDS buffer (no VGPR bounce)
#pragma unroll
      for (int k = 0; k < 4; ++k)
        __builtin_amdgcn_global_load_async_to_lds_b128(
            (v4i*)(src + k * 8), (v4i*)&As[nxt][aRow][aCol + k * 8], 0, 0);
#else
#pragma unroll
      for (int k = 0; k < 4; ++k) ar[k] = *(const v8bf*)(src + k * 8);
#endif
      const __bf16* xs = xBase + (long)(c0 + bC) * PHW + d;
#pragma unroll
      for (int j = 0; j < 16; ++j) br[j] = xs[q[j]];
    }
    // prefetch A tile two stages ahead (global_prefetch_b8)
    if (it + 2 < NIT) {
      const int it2 = it + 2;
      const int kv2 = it2 >> 2;
      const int c02 = (it2 & 3) * 64;
      __builtin_prefetch(wBase + (long)kv2 * (CO * CI) + (long)aRow * CI + c02 + aCol, 0, 1);
    }

    // ---- compute on LDS[cur]: 2 K-chunks x (2x2) WMMA = 8 per wave
#pragma unroll
    for (int ks = 0; ks < 2; ++ks) {
      union { v16bf v; v8bf h[2]; } af[2], bfr[2];
#pragma unroll
      for (int mi = 0; mi < 2; ++mi) {
        const __bf16* p = &As[cur][mBase + mi * 16][ks * 32 + ak0];
        af[mi].h[0] = *(const v8bf*)p;
        af[mi].h[1] = *(const v8bf*)(p + 16);
      }
#pragma unroll
      for (int ni = 0; ni < 2; ++ni) {
        const __bf16* p = &Bs[cur][nBase + ni * 16][ks * 32 + bk0];
        bfr[ni].h[0] = *(const v8bf*)p;
        bfr[ni].h[1] = *(const v8bf*)(p + 16);
      }
#pragma unroll
      for (int mi = 0; mi < 2; ++mi)
#pragma unroll
        for (int ni = 0; ni < 2; ++ni)
          acc[mi][ni] = __builtin_amdgcn_wmma_f32_16x16x32_bf16(
              false, af[mi].v, false, bfr[ni].v,
              (short)0, acc[mi][ni], false, false);
    }

    // ---- publish next stage; single barrier per stage
    if (it + 1 < NIT) {
#if !USE_ASYNC_LDS
#pragma unroll
      for (int k = 0; k < 4; ++k) *(v8bf*)&As[nxt][aRow][aCol + k * 8] = ar[k];
#endif
#pragma unroll
      for (int j = 0; j < 16; ++j) Bs[nxt][bN + j][bC] = br[j];
#if USE_ASYNC_LDS
      __builtin_amdgcn_s_wait_asynccnt(0);
#endif
      __syncthreads();
    }
  }

  // ---- epilogue: f32 C/D layout (lanes 0-15: M=r; 16-31: M=r+8; N=lane%16)
#pragma unroll
  for (int mi = 0; mi < 2; ++mi) {
#pragma unroll
    for (int ni = 0; ni < 2; ++ni) {
#pragma unroll
      for (int r = 0; r < 8; ++r) {
        int o = m0 + wm * 32 + mi * 16 + (lane >> 4) * 8 + r;
        int n = n0 + wn * 32 + ni * 16 + (lane & 15);
        out[((long)b * CO + o) * HW + n] = acc[mi][ni][r] + Bc[b * CO + o];
      }
    }
  }
}

// ---------------------------------------------------------------------------
// Fallback (only if workspace too small): naive direct conv, fp32 VALU
// ---------------------------------------------------------------------------
__global__ __launch_bounds__(256) void moe_conv_naive_kernel(
    const float* __restrict__ x, const float* __restrict__ att,
    const float* __restrict__ w, const float* __restrict__ bias,
    float* __restrict__ out) {
  long idx = (long)blockIdx.x * 256 + threadIdx.x;
  if (idx >= (long)NB * CO * HW) return;
  int n = (int)(idx % HW);
  int o = (int)((idx / HW) % CO);
  int b = (int)(idx / ((long)CO * HW));
  int oh = n / WW, ow = n % WW;
  float a[NE];
#pragma unroll
  for (int e = 0; e < NE; ++e) a[e] = att[b * NE + e];
  float s = 0.0f;
#pragma unroll
  for (int e = 0; e < NE; ++e) s += a[e] * bias[e * CO + o];
  for (int c = 0; c < CI; ++c) {
    const float* xr = x + ((long)b * CI + c) * HW;
    for (int kh = 0; kh < 3; ++kh) {
      int ih = oh + kh - 1;
      if ((unsigned)ih >= (unsigned)HH) continue;
      for (int kw = 0; kw < 3; ++kw) {
        int iw = ow + kw - 1;
        if ((unsigned)iw >= (unsigned)WW) continue;
        float wm = 0.0f;
#pragma unroll
        for (int e = 0; e < NE; ++e)
          wm += a[e] * w[(((long)e * CO + o) * CI + c) * 9 + kh * 3 + kw];
        s += xr[ih * WW + iw] * wm;
      }
    }
  }
  out[idx] = s;
}

// ---------------------------------------------------------------------------
extern "C" void kernel_launch(void* const* d_in, const int* in_sizes, int n_in,
                              void* d_out, int out_size, void* d_ws, size_t ws_size,
                              hipStream_t stream) {
  const float* x    = (const float*)d_in[0];   // [32,256,56,56]
  const float* att  = (const float*)d_in[1];   // [32,8]
  const float* w    = (const float*)d_in[2];   // [8,256,256,3,3]
  const float* bias = (const float*)d_in[3];   // [8,256]
  float* out = (float*)d_out;                  // [32,256,56,56]

  const size_t wcBytes = (size_t)NB * 9 * CO * CI * sizeof(__bf16);  // 37,748,736
  const size_t xpBytes = (size_t)NB * CI * PHW * sizeof(__bf16);     // 55,115,776
  const size_t bcBytes = (size_t)NB * CO * sizeof(float);            // 32,768
  const size_t need = wcBytes + xpBytes + bcBytes;

  if (ws_size >= need) {
    __bf16* Wc = (__bf16*)d_ws;
    __bf16* Xp = (__bf16*)((char*)d_ws + wcBytes);
    float*  Bc = (float*)((char*)d_ws + wcBytes + xpBytes);

    combine_w_kernel<<<73728, 256, 0, stream>>>(att, w, Wc);
    combine_bias_kernel<<<NB, CO, 0, stream>>>(att, bias, Bc);
    const long npad = (long)NB * CI * PHW;     // 27,557,888
    convert_pad_kernel<<<(int)((npad + 255) / 256), 256, 0, stream>>>(x, Xp);

    dim3 grid(49, 2, 32);
    moe_conv_wmma_kernel<<<grid, 256, 0, stream>>>(Wc, Xp, Bc, out);
  } else {
    long total = (long)NB * CO * HW;
    moe_conv_naive_kernel<<<(int)((total + 255) / 256), 256, 0, stream>>>(
        x, att, w, bias, out);
  }
}
